// GraphAttentionLayer_74929999446274
// MI455X (gfx1250) — compile-verified, matrix operand order checked by reading, not run on hardware
//
#include <hip/hip_runtime.h>
#include <hip/hip_bf16.h>
#include <math.h>

#define ALPHA 0.2f
#define NROW 8192
#define FIN  512
#define FOUT 64

typedef __attribute__((ext_vector_type(16))) __bf16 v16bf;
typedef __attribute__((ext_vector_type(8)))  float  v8f;

__device__ __forceinline__ float leaky(float v) { return v > 0.f ? v : ALPHA * v; }

// ---------------------------------------------------------------------------
// K1: Wh = x @ W  via v_wmma_f32_16x16x32_bf16.  One wave per 16x16 D tile.
// A (16x32 bf16) layout: lane holds row M=lane&15; VGPR v holds K pair
//   K0 = (v/4)*16 + (lane>>4)*8 + (v%4)*2  (ISA 16-bit A table).
// B (32x16 bf16): lane holds col N=lane&15, same K packing (B^T-like).
// ---------------------------------------------------------------------------
__global__ __launch_bounds__(256) void gat_wh_kernel(
    const float* __restrict__ x, const float* __restrict__ w,
    float* __restrict__ Wh)
{
  const int lane = threadIdx.x & 31;
  const int wave = threadIdx.x >> 5;
  const int tile = blockIdx.x * 8 + wave;      // 2048 tiles = 512 rt x 4 ct
  const int rt = tile >> 2;
  const int ct = tile & 3;
  const int half = lane >> 4;
  const int l15  = lane & 15;
  const int rrow = rt * 16 + l15;              // A row for this lane
  const int col  = ct * 16 + l15;              // B/D col for this lane

  v8f acc = {};
  for (int kk = 0; kk < FIN; kk += 32) {
    v16bf a, b;
#pragma unroll
    for (int v = 0; v < 8; ++v) {
      const int g = v >> 2, vv = v & 3;
      const int k0 = kk + g * 16 + half * 8 + vv * 2;
      const float2 xa = *(const float2*)(x + (size_t)rrow * FIN + k0);
      a[2 * v]     = (__bf16)xa.x;
      a[2 * v + 1] = (__bf16)xa.y;
      b[2 * v]     = (__bf16)w[(size_t)(k0)     * FOUT + col];
      b[2 * v + 1] = (__bf16)w[(size_t)(k0 + 1) * FOUT + col];
    }
    acc = __builtin_amdgcn_wmma_f32_16x16x32_bf16(
        false, a, false, b, (short)0, acc, false, false);
  }
  // C/D layout: VGPR r -> row (r + 8*half), col = lane&15 within tile
#pragma unroll
  for (int r = 0; r < 8; ++r)
    Wh[(size_t)(rt * 16 + r + 8 * half) * FOUT + col] = acc[r];
}

// ---------------------------------------------------------------------------
// K2: s1 = Wh @ a1, s2 = Wh @ a2  (one thread per row)
// ---------------------------------------------------------------------------
__global__ __launch_bounds__(256) void gat_scores_kernel(
    const float* __restrict__ Wh, const float* __restrict__ a,
    float* __restrict__ s1, float* __restrict__ s2)
{
  const int i = blockIdx.x * blockDim.x + threadIdx.x;
  if (i >= NROW) return;
  float r1 = 0.f, r2 = 0.f;
#pragma unroll 8
  for (int c = 0; c < FOUT; ++c) {
    const float v = Wh[(size_t)i * FOUT + c];
    r1 += v * a[c];
    r2 += v * a[FOUT + c];
  }
  s1[i] = r1;
  s2[i] = r2;
}

// ---------------------------------------------------------------------------
// K2b: repack Wh (fp32 row-major) into bf16 B-operand layout so K3's B loads
// are contiguous 32B per lane.  Linear out index o = ((jc*4+nt)*32+lane)*16+e.
// ---------------------------------------------------------------------------
__global__ __launch_bounds__(256) void gat_pack_kernel(
    const float* __restrict__ Wh, __bf16* __restrict__ WhB)
{
  const int o = blockIdx.x * blockDim.x + threadIdx.x;  // 0 .. 524287
  const int elem = o & 15;
  const int lane = (o >> 4) & 31;
  const int nt   = (o >> 9) & 3;
  const int jc   = o >> 11;                 // 32-row K chunk, 0..255
  const int half = lane >> 4;
  const int col  = nt * 16 + (lane & 15);
  const int v = elem >> 1, g = v >> 2, vv = v & 3;
  const int k0 = g * 16 + half * 8 + vv * 2 + (elem & 1);
  const int row = jc * 32 + k0;
  WhB[o] = (__bf16)Wh[(size_t)row * FOUT + col];
}

// ---------------------------------------------------------------------------
// K2c: global max of s2 (single block reduction)
// ---------------------------------------------------------------------------
__global__ __launch_bounds__(256) void gat_s2max_kernel(
    const float* __restrict__ s2, float* __restrict__ s2max)
{
  __shared__ float red[256];
  float m = -INFINITY;
  for (int i = threadIdx.x; i < NROW; i += 256) m = fmaxf(m, s2[i]);
  red[threadIdx.x] = m;
  __syncthreads();
  for (int s = 128; s > 0; s >>= 1) {
    if ((int)threadIdx.x < s)
      red[threadIdx.x] = fmaxf(red[threadIdx.x], red[threadIdx.x + s]);
    __syncthreads();
  }
  if (threadIdx.x == 0) *s2max = red[0];
}

// ---------------------------------------------------------------------------
// K3: fused masked softmax + attention@Wh + ELU.
// One block (8 waves) per 16-row tile.  Each wave streams j-chunks of 32,
// builds the p tile (16x32 bf16) directly in the A-operand layout, and does
// 4 WMMAs per chunk against pre-packed Wh B tiles.  No rescale needed since
// m_r is a precomputed upper bound, so partials combine by ds_add_f32.
// ---------------------------------------------------------------------------
__global__ __launch_bounds__(256) void gat_attn_kernel(
    const int* __restrict__ adj, const __bf16* __restrict__ WhB,
    const float* __restrict__ s1, const float* __restrict__ s2,
    const float* __restrict__ s2max_p, float* __restrict__ out)
{
  const int lane = threadIdx.x & 31;
  const int wave = threadIdx.x >> 5;
  const int half = lane >> 4;
  const int r    = lane & 15;              // A-layout row within tile
  const int i0   = blockIdx.x * 16;

  __shared__ float lds_acc[16 * FOUT];
  __shared__ float lds_l[16];
  for (int t = threadIdx.x; t < 16 * FOUT; t += 256) lds_acc[t] = 0.f;
  if (threadIdx.x < 16) lds_l[threadIdx.x] = 0.f;
  __syncthreads();

  const float s1v = s1[i0 + r];
  const float m_r = leaky(s1v + *s2max_p);   // safe per-row upper bound
  const int* adjrow = adj + (size_t)(i0 + r) * NROW;

  v8f acc0 = {}, acc1 = {}, acc2 = {}, acc3 = {};
  float lsum = 0.f;

  for (int chunk = wave; chunk < NROW / 32; chunk += 8) {
    const int j0 = chunk * 32;
    __builtin_prefetch(adjrow + j0 + 8 * 32, 0, 0);   // next chunk of adj

    // adj for this lane: two contiguous 8-int runs (g=0 and g=1 K groups)
    const int4 m0 = *(const int4*)(adjrow + j0 + half * 8);
    const int4 m1 = *(const int4*)(adjrow + j0 + half * 8 + 4);
    const int4 m2 = *(const int4*)(adjrow + j0 + 16 + half * 8);
    const int4 m3 = *(const int4*)(adjrow + j0 + 16 + half * 8 + 4);
    const int msk[16] = {m0.x, m0.y, m0.z, m0.w, m1.x, m1.y, m1.z, m1.w,
                         m2.x, m2.y, m2.z, m2.w, m3.x, m3.y, m3.z, m3.w};

    v16bf pA;
#pragma unroll
    for (int v = 0; v < 8; ++v) {
      const int g = v >> 2, vv = v & 3;
#pragma unroll
      for (int e = 0; e < 2; ++e) {
        const int k0 = g * 16 + half * 8 + vv * 2 + e;
        const int mi = g * 8 + vv * 2 + e;
        float ev = s1v + s2[j0 + k0];
        ev = leaky(ev);
        ev = (msk[mi] > 0) ? ev : -9.0e15f;
        const float p = __expf(ev - m_r);   // masked -> underflows to 0
        const __bf16 pb = (__bf16)p;
        pA[2 * v + e] = pb;
        lsum += (float)pb;                  // keep l consistent with bf16 p
      }
    }

    // B operands: contiguous 32B per lane, coalesced across the wave
    const __bf16* bp = WhB + ((size_t)chunk * 4 * 32 + lane) * 16;
    const v16bf b0 = *(const v16bf*)(bp);
    const v16bf b1 = *(const v16bf*)(bp + 32 * 16);
    const v16bf b2 = *(const v16bf*)(bp + 64 * 16);
    const v16bf b3 = *(const v16bf*)(bp + 96 * 16);

    acc0 = __builtin_amdgcn_wmma_f32_16x16x32_bf16(false, pA, false, b0, (short)0, acc0, false, false);
    acc1 = __builtin_amdgcn_wmma_f32_16x16x32_bf16(false, pA, false, b1, (short)0, acc1, false, false);
    acc2 = __builtin_amdgcn_wmma_f32_16x16x32_bf16(false, pA, false, b2, (short)0, acc2, false, false);
    acc3 = __builtin_amdgcn_wmma_f32_16x16x32_bf16(false, pA, false, b3, (short)0, acc3, false, false);
  }

  // combine partial (l, acc) across waves in LDS (pure sums; no rescale)
  atomicAdd(&lds_l[r], lsum);
#pragma unroll
  for (int rr = 0; rr < 8; ++rr) {
    const int row = rr + 8 * half;
    atomicAdd(&lds_acc[row * FOUT +  0 + (lane & 15)], acc0[rr]);
    atomicAdd(&lds_acc[row * FOUT + 16 + (lane & 15)], acc1[rr]);
    atomicAdd(&lds_acc[row * FOUT + 32 + (lane & 15)], acc2[rr]);
    atomicAdd(&lds_acc[row * FOUT + 48 + (lane & 15)], acc3[rr]);
  }
  __syncthreads();

  // finalize: h = acc / l, ELU, store
  for (int t = threadIdx.x; t < 16 * FOUT; t += 256) {
    const int row = t >> 6;
    const int colx = t & 63;
    float h = lds_acc[t] / lds_l[row];
    h = h > 0.f ? h : (__expf(h) - 1.0f);
    out[(size_t)(i0 + row) * FOUT + colx] = h;
  }
}

// ---------------------------------------------------------------------------
extern "C" void kernel_launch(void* const* d_in, const int* in_sizes, int n_in,
                              void* d_out, int out_size, void* d_ws, size_t ws_size,
                              hipStream_t stream) {
  const float* x    = (const float*)d_in[0];   // 8192 x 512
  const int*   adj  = (const int*)d_in[1];     // 8192 x 8192
  const float* w    = (const float*)d_in[2];   // 512 x 64
  const float* a    = (const float*)d_in[3];   // 128 x 1
  float* out = (float*)d_out;                  // 8192 x 64

  // workspace layout
  float*  Wh  = (float*)d_ws;                  // 8192*64 f32   (2 MB)
  float*  s1  = Wh + (size_t)NROW * FOUT;      // 8192 f32
  float*  s2  = s1 + NROW;                     // 8192 f32
  float*  s2m = s2 + NROW;                     // 1 f32 (+pad)
  __bf16* WhB = (__bf16*)(s2m + 16);           // 8192*64 bf16  (1 MB)

  gat_wh_kernel<<<2048 / 8, 256, 0, stream>>>(x, w, Wh);
  gat_scores_kernel<<<NROW / 256, 256, 0, stream>>>(Wh, a, s1, s2);
  gat_pack_kernel<<<(NROW * FOUT) / 256, 256, 0, stream>>>(Wh, WhB);
  gat_s2max_kernel<<<1, 256, 0, stream>>>(s2, s2m);
  gat_attn_kernel<<<NROW / 16, 256, 0, stream>>>(adj, WhB, s1, s2, s2m, out);
}
